// SlowFastSSMM_SampleLevel_85478439125026
// MI455X (gfx1250) — matrix operand order repaired and checked by reading, not compiled, not run on hardware
//
#include <hip/hip_runtime.h>
#include <hip/hip_bf16.h>

// ---------------- problem constants (from reference) ----------------
#define DELTA 16
#define L_S   32
#define HID   8
#define GH    64
#define G3    192          // 3*GH
#define OD    16           // 2*HID
#define Bsz   128
#define Tlen  65536
#define Fnum  4095         // (T - L_S)/DELTA + 1
#define SEQS  (Bsz*Fnum)   // 524160 GRU sequences
#define NTILES (SEQS/16)   // 32760 (exact multiple)
#define WAVES 4            // waves per block in GRU kernel
#define HSTR  72           // padded LDS row stride for h (in halves)

#define NCH   128          // fast-branch chunks
#define CLEN  (Tlen/NCH)   // 512 steps per chunk (multiple of DELTA)

typedef __attribute__((ext_vector_type(16))) _Float16 v16h;
typedef __attribute__((ext_vector_type(8)))  _Float16 v8h;
typedef __attribute__((ext_vector_type(8)))  float    v8f;

__device__ __forceinline__ float sigf(float v)      { return 1.f/(1.f+__expf(-v)); }
__device__ __forceinline__ float tanhfast(float v)  { return 1.f - 2.f/(__expf(2.f*v)+1.f); }

// 16 contiguous halves -> v16h (B-fragment: one w_hh row slice, lane = N column)
__device__ __forceinline__ v16h ldB16(const _Float16* p){
  v8h lo = *(const v8h*)p;
  v8h hh = *(const v8h*)(p+8);
  v16h r;
  #pragma unroll
  for (int j=0;j<8;++j){ r[j]=lo[j]; r[8+j]=hh[j]; }
  return r;
}

// ==================== Kernel 1: slow branch (GRU) via WMMA ====================
// One wave handles 16 sequences. Per step: gh = h(16x64) @ w_hh.T(64x192) with
// 24 v_wmma_f32_16x16x32_f16 (2 K-tiles x 12 N-tiles), gates fused elementwise
// in the D layout, h round-trips through LDS to re-enter as an A-fragment.
__global__ __launch_bounds__(WAVES*32)
void gru_wmma_kernel(const float* __restrict__ x,
                     const float* __restrict__ w_ih, const float* __restrict__ w_hh,
                     const float* __restrict__ b_ih, const float* __restrict__ b_hh,
                     const float* __restrict__ w_sl, const float* __restrict__ b_sl,
                     float* __restrict__ epsb)
{
  __shared__ __align__(16) _Float16 s_whh[G3*GH];          // [gate g][k] row-major f16
  __shared__ __align__(16) _Float16 s_wsl[OD*GH];          // [o][k] f16
  __shared__ __align__(16) _Float16 s_h[WAVES][16*HSTR];   // per-wave h, 16 x 64 (padded)
  __shared__ float s_xf[WAVES][L_S*16];                    // frames [t][m]

  const int lane = threadIdx.x & 31;
  const int wv   = threadIdx.x >> 5;
  const int c    = lane & 15;   // matrix column (N) / A row (M)
  const int hi   = lane >> 4;   // lane half
  const int tile = blockIdx.x * WAVES + wv;

  // block-cooperative weight staging (f32 -> f16)
  for (int i = threadIdx.x; i < G3*GH; i += blockDim.x) s_whh[i] = (_Float16)w_hh[i];
  for (int i = threadIdx.x; i < OD*GH; i += blockDim.x) s_wsl[i] = (_Float16)w_sl[i];

  if (tile < NTILES) {
    for (int i = lane; i < 16*HSTR; i += 32) s_h[wv][i] = (_Float16)0.f;
    for (int i = lane; i < 16*L_S; i += 32) {
      int m = i & 15, t = i >> 4;
      int seq = tile*16 + m;
      int b = seq / Fnum, f = seq - b*Fnum;
      s_xf[wv][t*16 + m] = x[(size_t)b*Tlen + (size_t)f*DELTA + t];
    }
  }
  __syncthreads();
  if (tile >= NTILES) return;   // wave-uniform: EXEC stays all-ones for WMMA

  // per-lane gate constants: this lane owns gate columns g = ht*16+c (+64,+128)
  float wr[4],wz[4],wn[4],br[4],bz[4],bn[4],cr[4],cz[4],cn[4];
  #pragma unroll
  for (int ht=0; ht<4; ++ht){
    int g = ht*16 + c;
    wr[ht]=w_ih[g]; wz[ht]=w_ih[g+GH]; wn[ht]=w_ih[g+2*GH];
    br[ht]=b_ih[g]; bz[ht]=b_ih[g+GH]; bn[ht]=b_ih[g+2*GH];
    cr[ht]=b_hh[g]; cz[ht]=b_hh[g+GH]; cn[ht]=b_hh[g+2*GH];
  }
  float hreg[4][8];             // h kept in f32 in the D layout
  #pragma unroll
  for (int ht=0;ht<4;++ht)
    #pragma unroll
    for (int e=0;e<8;++e) hreg[ht][e]=0.f;

  for (int t=0; t<L_S; ++t){
    // cross-lane LDS RAW: h stores from previous step -> A-frag loads now
    asm volatile("s_wait_dscnt 0" ::: "memory");

    // A-fragments (contiguous k = 16*hi + s, same order as the B loads): low lanes K=0..15; high lanes K=16..31
    v16h a0, a1;
    {
      const _Float16* hp = &s_h[wv][c*HSTR];
      v8h p0 = *(const v8h*)(hp +      16*hi);
      v8h p1 = *(const v8h*)(hp +  8 + 16*hi);
      v8h p2 = *(const v8h*)(hp + 32 + 16*hi);
      v8h p3 = *(const v8h*)(hp + 40 + 16*hi);
      #pragma unroll
      for (int j=0;j<8;++j){ a0[j]=p0[j]; a0[8+j]=p1[j]; a1[j]=p2[j]; a1[8+j]=p3[j]; }
    }
    float xv[8];
    #pragma unroll
    for (int e=0;e<8;++e) xv[e] = s_xf[wv][t*16 + e + 8*hi];

    #pragma unroll
    for (int ht=0; ht<4; ++ht){
      const int gr = ht*16 + c;
      v8f accR = {}; v8f accZ = {}; v8f accN = {};
      #pragma unroll
      for (int kt=0; kt<2; ++kt){
        const int kof = kt*32 + hi*16;
        v16h bR = ldB16(&s_whh[(size_t)(gr        )*GH + kof]);
        v16h bZ = ldB16(&s_whh[(size_t)(gr +   GH )*GH + kof]);
        v16h bN = ldB16(&s_whh[(size_t)(gr + 2*GH )*GH + kof]);
        v16h av = kt ? a1 : a0;
        accR = __builtin_amdgcn_wmma_f32_16x16x32_f16(false, av, false, bR, (short)0, accR, false, false);
        accZ = __builtin_amdgcn_wmma_f32_16x16x32_f16(false, av, false, bZ, (short)0, accZ, false, false);
        accN = __builtin_amdgcn_wmma_f32_16x16x32_f16(false, av, false, bN, (short)0, accN, false, false);
      }
      // gate fusion: D layout => (row m = e+8*hi, col = gate c) purely elementwise
      #pragma unroll
      for (int e=0;e<8;++e){
        float xt = xv[e];
        float rg = sigf(xt*wr[ht] + br[ht] + accR[e] + cr[ht]);
        float zg = sigf(xt*wz[ht] + bz[ht] + accZ[e] + cz[ht]);
        float ng = tanhfast(xt*wn[ht] + bn[ht] + rg*(accN[e] + cn[ht]));
        float hn = (1.f - zg)*ng + zg*hreg[ht][e];
        hreg[ht][e] = hn;
        s_h[wv][(e + 8*hi)*HSTR + ht*16 + c] = (_Float16)hn;   // back to A-side layout
      }
    }
  }

  // eps = hT @ w_sl.T + b_sl  (one 16x16 tile, 2 K-tiles)
  asm volatile("s_wait_dscnt 0" ::: "memory");
  v16h a0, a1;
  {
    const _Float16* hp = &s_h[wv][c*HSTR];
    v8h p0 = *(const v8h*)(hp +      16*hi);
    v8h p1 = *(const v8h*)(hp +  8 + 16*hi);
    v8h p2 = *(const v8h*)(hp + 32 + 16*hi);
    v8h p3 = *(const v8h*)(hp + 40 + 16*hi);
    #pragma unroll
    for (int j=0;j<8;++j){ a0[j]=p0[j]; a0[8+j]=p1[j]; a1[j]=p2[j]; a1[8+j]=p3[j]; }
  }
  v8f acc = {};
  #pragma unroll
  for (int kt=0; kt<2; ++kt){
    v16h bS = ldB16(&s_wsl[(size_t)c*GH + kt*32 + hi*16]);
    v16h av = kt ? a1 : a0;
    acc = __builtin_amdgcn_wmma_f32_16x16x32_f16(false, av, false, bS, (short)0, acc, false, false);
  }
  float bs = b_sl[c];
  #pragma unroll
  for (int e=0;e<8;++e){
    int seq = tile*16 + e + 8*hi;
    epsb[(size_t)seq*OD + c] = acc[e] + bs;
  }
}

// ==================== Kernel 2a: chunk summaries (P = prod A, S = scan) ====================
__global__ __launch_bounds__(256)
void fast_partial_kernel(const float* __restrict__ x, const float* __restrict__ epsb,
                         const float* __restrict__ w_in,
                         float* __restrict__ Pc, float* __restrict__ Sc)
{
  int idx = blockIdx.x*blockDim.x + threadIdx.x;
  if (idx >= Bsz*NCH) return;
  int b = idx / NCH, ch = idx - b*NCH;
  float win[HID];
  #pragma unroll
  for (int k=0;k<HID;++k) win[k]=w_in[k];
  float p[HID], s[HID];
  #pragma unroll
  for (int k=0;k<HID;++k){ p[k]=1.f; s[k]=0.f; }
  const int t0 = ch*CLEN;
  const float* xb = x + (size_t)b*Tlen;
  for (int tb=0; tb<CLEN; tb+=DELTA){
    int t = t0 + tb;
    int j = t/DELTA - 1; j = j<0 ? 0 : (j>Fnum-1 ? Fnum-1 : j);
    const float* er = epsb + (size_t)(b*Fnum + j)*OD;
    float A[HID], g[HID], A16[HID];
    #pragma unroll
    for (int k=0;k<HID;++k){ A[k]=sigf(er[k]); g[k]=er[HID+k]; }
    #pragma unroll
    for (int k=0;k<HID;++k){ float a2=A[k]*A[k], a4=a2*a2, a8=a4*a4; A16[k]=a8*a8; }
    __builtin_prefetch(xb + t + 4*DELTA, 0, 0);   // global_prefetch_b8
    #pragma unroll 4
    for (int u=0;u<DELTA;++u){
      float xt = xb[t+u];
      #pragma unroll
      for (int k=0;k<HID;++k) s[k] = A[k]*s[k] + xt*win[k]*g[k];
    }
    #pragma unroll
    for (int k=0;k<HID;++k) p[k]*=A16[k];
  }
  #pragma unroll
  for (int k=0;k<HID;++k){ Pc[(size_t)idx*HID+k]=p[k]; Sc[(size_t)idx*HID+k]=s[k]; }
}

// ==================== Kernel 2b: tiny sequential scan over chunk summaries ====================
__global__ __launch_bounds__(256)
void fast_scan_kernel(const float* __restrict__ Pc, const float* __restrict__ Sc,
                      float* __restrict__ Hin)
{
  int idx = blockIdx.x*blockDim.x + threadIdx.x;
  if (idx >= Bsz*HID) return;
  int b = idx / HID, k = idx - b*HID;
  float h = 0.f;
  for (int ch=0; ch<NCH; ++ch){
    size_t o = ((size_t)(b*NCH + ch))*HID + k;
    Hin[o] = h;
    h = Pc[o]*h + Sc[o];
  }
}

// ==================== Kernel 2c: recompute with correct h_in, emit y ====================
__global__ __launch_bounds__(256)
void fast_final_kernel(const float* __restrict__ x, const float* __restrict__ epsb,
                       const float* __restrict__ w_in, const float* __restrict__ w_out,
                       const float* __restrict__ Hin, float* __restrict__ y)
{
  int idx = blockIdx.x*blockDim.x + threadIdx.x;
  if (idx >= Bsz*NCH) return;
  int b = idx / NCH, ch = idx - b*NCH;
  float win[HID], wo[HID], h[HID];
  #pragma unroll
  for (int k=0;k<HID;++k){ win[k]=w_in[k]; wo[k]=w_out[k]; h[k]=Hin[(size_t)idx*HID+k]; }
  const int t0 = ch*CLEN;
  const float* xb = x + (size_t)b*Tlen;
  float*       yb = y + (size_t)b*Tlen;
  for (int tb=0; tb<CLEN; tb+=DELTA){
    int t = t0 + tb;
    int j = t/DELTA - 1; j = j<0 ? 0 : (j>Fnum-1 ? Fnum-1 : j);
    const float* er = epsb + (size_t)(b*Fnum + j)*OD;
    float A[HID], g[HID];
    #pragma unroll
    for (int k=0;k<HID;++k){ A[k]=sigf(er[k]); g[k]=er[HID+k]; }
    __builtin_prefetch(xb + t + 4*DELTA, 0, 0);
    #pragma unroll 4
    for (int u=0;u<DELTA;++u){
      float xt = xb[t+u];
      float acc = 0.f;
      #pragma unroll
      for (int k=0;k<HID;++k){ h[k] = A[k]*h[k] + xt*win[k]*g[k]; acc += h[k]*wo[k]; }
      yb[t+u] = acc;
    }
  }
}

// ==================== launcher ====================
extern "C" void kernel_launch(void* const* d_in, const int* in_sizes, int n_in,
                              void* d_out, int out_size, void* d_ws, size_t ws_size,
                              hipStream_t stream) {
  const float* x    = (const float*)d_in[0];
  const float* w_in = (const float*)d_in[1];
  const float* w_out= (const float*)d_in[2];
  const float* w_ih = (const float*)d_in[3];
  const float* w_hh = (const float*)d_in[4];
  const float* b_ih = (const float*)d_in[5];
  const float* b_hh = (const float*)d_in[6];
  const float* w_sl = (const float*)d_in[7];
  const float* b_sl = (const float*)d_in[8];
  float* y = (float*)d_out;

  // workspace layout: eps (B*F x 16 f32, ~32 MB) | P | S | Hin (0.5 MB each)
  char* ws = (char*)d_ws;
  float* epsb = (float*)ws;
  size_t off  = (size_t)SEQS*OD*sizeof(float);
  float* Pc  = (float*)(ws + off); off += (size_t)Bsz*NCH*HID*sizeof(float);
  float* Sc  = (float*)(ws + off); off += (size_t)Bsz*NCH*HID*sizeof(float);
  float* Hin = (float*)(ws + off);

  gru_wmma_kernel<<<NTILES/WAVES, WAVES*32, 0, stream>>>(x, w_ih, w_hh, b_ih, b_hh, w_sl, b_sl, epsb);

  int n2 = Bsz*NCH;
  fast_partial_kernel<<<(n2+255)/256, 256, 0, stream>>>(x, epsb, w_in, Pc, Sc);
  fast_scan_kernel<<<(Bsz*HID+255)/256, 256, 0, stream>>>(Pc, Sc, Hin);
  fast_final_kernel<<<(n2+255)/256, 256, 0, stream>>>(x, epsb, w_in, w_out, Hin, y);
}
